// LSTM_MASK_84430467105161
// MI455X (gfx1250) — matrix operand order repaired and checked by reading, not compiled, not run on hardware
//
#include <hip/hip_runtime.h>
#include <hip/hip_bf16.h>

#define TT 1024
#define BB 512
#define II 128
#define HH 256
#define GG 1024 /* 4*H */
#define LDS_STRIDE 264 /* 256 + 8 halfs pad -> conflict-reduced ds_load_b128 */

typedef __attribute__((ext_vector_type(16))) _Float16 v16h;
typedef __attribute__((ext_vector_type(8)))  _Float16 v8h;
typedef __attribute__((ext_vector_type(8)))  float    v8f;
typedef __attribute__((ext_vector_type(4)))  float    f32x4;

__device__ __forceinline__ float fsigmoid(float x) {
    return 1.0f / (1.0f + __expf(-x));
}
__device__ __forceinline__ float ftanh(float x) {
    return 2.0f / (1.0f + __expf(-2.0f * x)) - 1.0f; // tanh(x)=2*sigmoid(2x)-1
}

// A fragment (16x32 f16): lanes 0-15 hold K {kk..kk+7, kk+16..kk+23},
// lanes 16-31 hold K {kk+8..kk+15, kk+24..kk+31}. Works for global or LDS rows.
__device__ __forceinline__ v16h load_a_f16(const _Float16* row, int kk, int hi) {
    const _Float16* p0 = row + kk + hi * 8;
    v8h lo = *(const v8h*)p0;
    v8h hv = *(const v8h*)(p0 + 16);
    v16h a;
#pragma unroll
    for (int e = 0; e < 8; ++e) { a[e] = lo[e]; a[8 + e] = hv[e]; }
    return a;
}

// Same A fragment sourced from an f32 row (converted in-register to f16).
__device__ __forceinline__ v16h load_a_f32(const float* __restrict__ row, int kk, int hi) {
    const float* p0 = row + kk + hi * 8;
    f32x4 a0 = *(const f32x4*)(p0 + 0);
    f32x4 a1 = *(const f32x4*)(p0 + 4);
    f32x4 b0 = *(const f32x4*)(p0 + 16);
    f32x4 b1 = *(const f32x4*)(p0 + 20);
    v16h a;
#pragma unroll
    for (int e = 0; e < 4; ++e) {
        a[e]      = (_Float16)a0[e];
        a[4 + e]  = (_Float16)a1[e];
        a[8 + e]  = (_Float16)b0[e];
        a[12 + e] = (_Float16)b1[e];
    }
    return a;
}

// Init: f32->f16 weight conversion, fused bias, mask.
__global__ __launch_bounds__(256) void lstm_init(
    const float* __restrict__ inputs, const float* __restrict__ W_ih,
    const float* __restrict__ W_hh, const float* __restrict__ b_ih,
    const float* __restrict__ b_hh, _Float16* __restrict__ Wih_h,
    _Float16* __restrict__ Whh_h, float* __restrict__ bias,
    float* __restrict__ maskbuf)
{
    int id = blockIdx.x * blockDim.x + threadIdx.x; // 0 .. 524287
    if (id < GG * HH) Whh_h[id] = (_Float16)W_hh[id];
    if (id < GG * II) Wih_h[id] = (_Float16)W_ih[id];
    if (id < GG)      bias[id]  = b_ih[id] + b_hh[id];
    if (id < TT * BB) maskbuf[id] = (inputs[(size_t)id * II + (II - 1)] > 0.0f) ? 1.0f : 0.0f;
}

// Persistent full-sequence scan. One workgroup (16 waves) owns a 16-row batch
// tile and ALL hidden columns: the recurrence never crosses workgroups, so the
// entire T loop runs in one launch. h lives in double-buffered LDS, c in VGPRs.
// Weights are re-read from L2 each step; a per-iteration opaque zero offset
// (asm barrier on an SGPR integer, NOT on the pointer — pointer provenance is
// preserved so loads stay global_load_b128, not flat) stops LICM from hoisting
// 384 VGPRs of fragments out of the loop and spilling to scratch.
__global__ __launch_bounds__(512, 1) void lstm_scan(
    const float* __restrict__ inputs,      // [T,B,I] f32
    const _Float16* __restrict__ Wih_h,    // [4H,I]
    const _Float16* __restrict__ Whh_h,    // [4H,H]
    const float* __restrict__ bias,        // [4H]
    const float* __restrict__ maskbuf,     // [T,B]
    float* __restrict__ hlast)             // [B,H] unmasked h at t=T-1
{
    __shared__ _Float16 hsh[2][16][LDS_STRIDE];

    const int lane = threadIdx.x & 31;
    const int jt   = threadIdx.x >> 5;     // wave id == hidden tile, 0..15
    const int b0   = blockIdx.x * 16;      // batch tile (grid = 32)
    const int j0   = jt * 16;
    const int n    = lane & 15;
    const int hi   = lane >> 4;

    // zero initial h buffer (buffer 0)
    for (int i = threadIdx.x; i < 16 * LDS_STRIDE; i += 512)
        hsh[0][0][i] = (_Float16)0.0f;

    float bq[4];
#pragma unroll
    for (int q = 0; q < 4; ++q) bq[q] = bias[q * HH + j0 + n];

    float creg[8];
#pragma unroll
    for (int r = 0; r < 8; ++r) creg[r] = 0.0f;

    __syncthreads();

#pragma unroll 1
    for (int t = 0; t < TT; ++t) {
        // Opaque zero: redefined by volatile asm each iteration so every weight
        // address depends on it -> fragment loads can't be hoisted/CSE'd across
        // the scan loop, yet address-space inference still sees global pointers.
        int zoff = 0;
        asm volatile("" : "+s"(zoff));

        v8f acc[4];
#pragma unroll
        for (int q = 0; q < 4; ++q)
#pragma unroll
            for (int r = 0; r < 8; ++r) acc[q][r] = bq[q];

        const float* xrow = inputs + ((size_t)t * BB + b0 + n) * II;

        // prefetch next timestep's input slice while this step computes
        if (t + 1 < TT) {
            const char* pf =
                (const char*)(inputs + ((size_t)(t + 1) * BB + b0 + n) * II) + hi * 256;
            __builtin_prefetch(pf, 0, 0);
            __builtin_prefetch(pf + 128, 0, 0);
        }

        // ---- input projection: K = I = 128 (A from global f32) ----
#pragma unroll
        for (int kk = 0; kk < II; kk += 32) {
            v16h a = load_a_f32(xrow, kk, hi);
#pragma unroll
            for (int q = 0; q < 4; ++q) {
                v16h b = load_a_f16(Wih_h +
                    ((size_t)(q * HH + j0 + n) * II + zoff), kk, hi);
                acc[q] = __builtin_amdgcn_wmma_f32_16x16x32_f16(
                    false, a, false, b, (short)0, acc[q], false, false);
            }
        }

        // ---- recurrent projection: K = H = 256 (A from LDS f16) ----
        const _Float16* hrow = &hsh[t & 1][n][0];
#pragma unroll
        for (int kk = 0; kk < HH; kk += 32) {
            v16h a = load_a_f16(hrow, kk, hi);
#pragma unroll
            for (int q = 0; q < 4; ++q) {
                v16h b = load_a_f16(Whh_h +
                    ((size_t)(q * HH + j0 + n) * HH + zoff), kk, hi);
                acc[q] = __builtin_amdgcn_wmma_f32_16x16x32_f16(
                    false, a, false, b, (short)0, acc[q], false, false);
            }
        }

        // ---- LSTM elementwise (C/D layout: M = r + hi*8, N = lane&15) ----
#pragma unroll
        for (int r = 0; r < 8; ++r) {
            int m = r + hi * 8;
            int b = b0 + m;
            float mval = maskbuf[(size_t)t * BB + b];
            float iv = fsigmoid(acc[0][r]);
            float fv = fsigmoid(acc[1][r]);
            float gv = ftanh(acc[2][r]);
            float ov = fsigmoid(acc[3][r]);
            float c_new = fv * creg[r] + iv * gv;
            float h_new = ov * ftanh(c_new);
            creg[r] = mval * c_new;                                 // masked carry c (registers)
            hsh[(t + 1) & 1][m][j0 + n] = (_Float16)(mval * h_new); // masked carry h (LDS)
            if (t == TT - 1)
                hlast[(size_t)b * HH + j0 + n] = h_new;             // unmasked hs[T-1]
        }
        __syncthreads(); // next-step reads of buffer (t+1)&1 wait for all writes
    }
}

// likelihood[b] = hlast[b,:] . W_out + b_out ; one wave per batch row.
__global__ __launch_bounds__(256) void lstm_out(
    const float* __restrict__ hlast, const float* __restrict__ W_out,
    const float* __restrict__ b_out, float* __restrict__ out)
{
    int wave = blockIdx.x * 8 + (threadIdx.x >> 5);
    int lane = threadIdx.x & 31;
    const float* hr = hlast + (size_t)wave * HH;
    float s = 0.0f;
#pragma unroll
    for (int j = lane; j < HH; j += 32) s += hr[j] * W_out[j];
#pragma unroll
    for (int off = 16; off > 0; off >>= 1) s += __shfl_down(s, off, 32);
    if (lane == 0) out[wave] = s + b_out[0];
}

extern "C" void kernel_launch(void* const* d_in, const int* in_sizes, int n_in,
                              void* d_out, int out_size, void* d_ws, size_t ws_size,
                              hipStream_t stream) {
    const float* inputs = (const float*)d_in[0]; // [T,B,I]
    const float* W_ih   = (const float*)d_in[1]; // [4H,I]
    const float* W_hh   = (const float*)d_in[2]; // [4H,H]
    const float* b_ih   = (const float*)d_in[3]; // [4H]
    const float* b_hh   = (const float*)d_in[4]; // [4H]
    const float* W_out  = (const float*)d_in[5]; // [1,H]
    const float* b_out  = (const float*)d_in[6]; // [1]
    float* out = (float*)d_out;                  // [B]

    char* ws = (char*)d_ws;
    // workspace layout (~3.4 MB)
    _Float16* Whh_h = (_Float16*)(ws + 0);         // 524288 B
    _Float16* Wih_h = (_Float16*)(ws + 524288);    // 262144 B
    float*    bias  = (float*)   (ws + 786432);    // 4096 B
    float*    maskb = (float*)   (ws + 790528);    // 2097152 B
    float*    hlast = (float*)   (ws + 2887680);   // 524288 B

    // 1) weight conversion + fused bias + mask
    lstm_init<<<(TT * BB + 255) / 256, 256, 0, stream>>>(
        inputs, W_ih, W_hh, b_ih, b_hh, Wih_h, Whh_h, bias, maskb);

    // 2) entire 1024-step scan in ONE persistent launch
    //    (32 independent batch tiles, h via LDS, c in registers)
    lstm_scan<<<32, 512, 0, stream>>>(inputs, Wih_h, Whh_h, bias, maskb, hlast);

    // 3) output projection
    lstm_out<<<64, 256, 0, stream>>>(hlast, W_out, b_out, out);
}
